// VSDGCRNN_59253368815848
// MI455X (gfx1250) — matrix operand order repaired and checked by reading, not compiled, not run on hardware
//
#include <hip/hip_runtime.h>
#include <hip/hip_bf16.h>

// ---------------------------------------------------------------------------
// VSDGCRNN on MI455X (gfx1250): persistent per-batch workgroups, f16 WMMA
// GEMMs for msg + gates (q-factored), LDS-resident hidden state.
// B=64 T=128 N=64 D=64 Q=5; padded feature K: 129 -> 160.
// ---------------------------------------------------------------------------

#define BB 64
#define TT 128
#define NN 64
#define DD 64
#define QQ 5
#define FPAD 160
#define FREAL 129

// strides (in half elements) chosen so every fragment run is 16B aligned
#define CNF_S 168   // combined row-major   [n][f]
#define CT_S  72    // combined transposed  [f][n]
#define AB_S  72    // adjacency f16        [i][j]
#define MSG_S 168   // msg f16              [n][f]

typedef _Float16 v16h __attribute__((ext_vector_type(16)));
typedef _Float16 v8h  __attribute__((ext_vector_type(8)));
typedef float    v8f  __attribute__((ext_vector_type(8)));

// ---- workspace layout (float offsets / byte offsets) ----------------------
#define WS_QV    0        // 320 f32
#define WS_ADJ   512      // 4096 f32 (adj with zeroed diagonal)
#define WS_VT    4608     // 4096 f32 (var_total[b][n])
#define WS_BR    8704     // 4096 f32 bias r (per n,o)
#define WS_BU    12800    // 4096 f32 bias u
#define WS_BC    16896    // 4096 f32 bias c
#define WS_HID   20992    // 64*256 f32 MLP hidden (f | g)
#define WS_HALF_BYTE 149504  // = (20992+16384)*4 ; f16 weights start here
#define WTRU_N   102400   // 5*128*160 halves (r|u fused along output dim)
#define WTC_N    51200    // 5*64*160 halves

// ---------------------------------------------------------------------------
// fragment helpers (CDNA5 16-bit WMMA VGPR layouts)
// ---------------------------------------------------------------------------
__device__ __forceinline__ v16h mk16(v8h lo, v8h hi) {
  v16h r;
#pragma unroll
  for (int i = 0; i < 8; ++i) { r[i] = lo[i]; r[i + 8] = hi[i]; }
  return r;
}

// A-matrix 16x32 f16: lane L holds row m=m0+(L&15); K runs:
//   elems 0..15 -> K = k0 + (L>>4)*16 + 0..15
__device__ __forceinline__ v16h afrag(const _Float16* base, int stride,
                                      int m0, int k0, int lane) {
  const _Float16* p = base + (m0 + (lane & 15)) * stride + k0 + ((lane >> 4) << 4);
  return mk16(*(const v8h*)p, *(const v8h*)(p + 8));
}

// B-matrix 32x16 f16: lane L holds col n, elems e -> K = (L>>4)*16 + e,
// i.e. 16 contiguous halves starting at p (caller folds n and (L>>4)*16 in).
__device__ __forceinline__ v16h bfrag(const _Float16* p) {
  return mk16(*(const v8h*)p, *(const v8h*)(p + 8));
}

__device__ __forceinline__ v8f wmma32(v16h a, v16h b, v8f c) {
  return __builtin_amdgcn_wmma_f32_16x16x32_f16(false, a, false, b,
                                                (short)0, c, false, false);
}

__device__ __forceinline__ float sigm(float x) { return 1.0f / (1.0f + __expf(-x)); }

// ---------------------------------------------------------------------------
// P1a: hidden = relu(plm @ W1 + b1) for both MLPs (f and g), WG per node row
// ---------------------------------------------------------------------------
__global__ __launch_bounds__(256) void vsd_p1a(const float* __restrict__ plm,
                                               const float* __restrict__ Wf1,
                                               const float* __restrict__ bf1,
                                               const float* __restrict__ Wg1,
                                               const float* __restrict__ bg1,
                                               float* __restrict__ hid) {
  __shared__ float row[768];
  const int n = blockIdx.x, tid = threadIdx.x;
  for (int k = tid; k < 768; k += 256) row[k] = plm[n * 768 + k];
  __syncthreads();
  if (tid < 128) {
    float acc = bf1[tid];
    for (int k = 0; k < 768; ++k) acc += row[k] * Wf1[k * 128 + tid];
    hid[n * 256 + tid] = fmaxf(acc, 0.0f);
  } else {
    const int j = tid - 128;
    float acc = bg1[j];
    for (int k = 0; k < 768; ++k) acc += row[k] * Wg1[k * 128 + j];
    hid[n * 256 + 128 + j] = fmaxf(acc, 0.0f);
  }
}

// ---------------------------------------------------------------------------
// P1b: qv, ne (normalized), adj = softmax(ne ne^T) with zeroed diagonal
// ---------------------------------------------------------------------------
__global__ __launch_bounds__(256) void vsd_p1b(const float* __restrict__ hid,
                                               const float* __restrict__ Wf2,
                                               const float* __restrict__ bf2,
                                               const float* __restrict__ Wg2,
                                               const float* __restrict__ bg2,
                                               float* __restrict__ qv,
                                               float* __restrict__ adj) {
  __shared__ float ne[NN * 8];
  const int tid = threadIdx.x;
  for (int idx = tid; idx < NN * QQ; idx += 256) {
    const int n = idx / QQ, q = idx % QQ;
    float acc = bf2[q];
    for (int k = 0; k < 128; ++k) acc += hid[n * 256 + k] * Wf2[k * QQ + q];
    qv[idx] = acc;
  }
  for (int idx = tid; idx < NN * 8; idx += 256) {
    const int n = idx / 8, e = idx % 8;
    float acc = bg2[e];
    for (int k = 0; k < 128; ++k) acc += hid[n * 256 + 128 + k] * Wg2[k * 8 + e];
    ne[idx] = acc;
  }
  __syncthreads();
  if (tid < NN) {
    float s = 0.0f;
    for (int e = 0; e < 8; ++e) s += ne[tid * 8 + e] * ne[tid * 8 + e];
    const float inv = rsqrtf(s);
    for (int e = 0; e < 8; ++e) ne[tid * 8 + e] *= inv;
  }
  __syncthreads();
  if (tid < NN) {
    float sr[NN];
    float mx = -1e30f;
    for (int j = 0; j < NN; ++j) {
      float d = 0.0f;
      for (int e = 0; e < 8; ++e) d += ne[tid * 8 + e] * ne[j * 8 + e];
      sr[j] = d;
      mx = fmaxf(mx, d);
    }
    float sum = 0.0f;
    for (int j = 0; j < NN; ++j) { sr[j] = __expf(sr[j] - mx); sum += sr[j]; }
    const float inv = 1.0f / sum;
    for (int j = 0; j < NN; ++j)
      adj[tid * NN + j] = (j == tid) ? 0.0f : sr[j] * inv;
  }
}

// ---------------------------------------------------------------------------
// P2: var_total[b][n] = sum_t mask[b,t,n]
// ---------------------------------------------------------------------------
__global__ __launch_bounds__(256) void vsd_p2(const float* __restrict__ mask,
                                              float* __restrict__ vt) {
  const int idx = blockIdx.x * 256 + threadIdx.x;   // 4096
  const int b = idx >> 6, n = idx & 63;
  float s = 0.0f;
  for (int t = 0; t < TT; ++t) s += mask[(b * TT + t) * NN + n];
  vt[idx] = s;
}

// ---------------------------------------------------------------------------
// P3: gate weights -> f16, padded K 129->160, r|u fused, layout [q][o][k]
// ---------------------------------------------------------------------------
__global__ __launch_bounds__(256) void vsd_p3(const float* __restrict__ Wr,
                                              const float* __restrict__ Wu,
                                              const float* __restrict__ Wc,
                                              _Float16* __restrict__ wtru,
                                              _Float16* __restrict__ wtc) {
  const int idx = blockIdx.x * 256 + threadIdx.x;   // 153600 total
  if (idx < WTRU_N) {
    const int q = idx / (128 * FPAD);
    const int r = idx % (128 * FPAD);
    const int o = r / FPAD, k = r % FPAD;
    float v = 0.0f;
    if (k < FREAL)
      v = (o < 64) ? Wr[(q * FREAL + k) * 64 + o] : Wu[(q * FREAL + k) * 64 + (o - 64)];
    wtru[idx] = (_Float16)v;
  } else if (idx < WTRU_N + WTC_N) {
    const int j = idx - WTRU_N;
    const int q = j / (64 * FPAD);
    const int r = j % (64 * FPAD);
    const int o = r / FPAD, k = r % FPAD;
    const float v = (k < FREAL) ? Wc[(q * FREAL + k) * 64 + o] : 0.0f;
    wtc[j] = (_Float16)v;
  }
}

// ---------------------------------------------------------------------------
// P4: per-node biases b*_n = qv @ b*
// ---------------------------------------------------------------------------
__global__ __launch_bounds__(256) void vsd_p4(const float* __restrict__ qv,
                                              const float* __restrict__ br,
                                              const float* __restrict__ bu,
                                              const float* __restrict__ bc,
                                              float* __restrict__ biasr,
                                              float* __restrict__ biasu,
                                              float* __restrict__ biasc) {
  const int idx = blockIdx.x * 256 + threadIdx.x;   // 12288
  const int g = idx >> 12, r = idx & 4095;
  const int n = r >> 6, o = r & 63;
  const float* src = (g == 0) ? br : (g == 1) ? bu : bc;
  float acc = 0.0f;
  for (int q = 0; q < QQ; ++q) acc += qv[n * QQ + q] * src[q * 64 + o];
  ((g == 0) ? biasr : (g == 1) ? biasu : biasc)[r] = acc;
}

// ---------------------------------------------------------------------------
// Main: one WG per batch, persistent over T; three WMMA GEMM phases per step
// ---------------------------------------------------------------------------
__global__ __launch_bounds__(256) void vsd_main(
    const float* __restrict__ obs, const float* __restrict__ mask,
    const int* __restrict__ lengths, const float* __restrict__ avgint,
    const float* __restrict__ rarity, const float* __restrict__ adjnoI,
    const float* __restrict__ vt, const float* __restrict__ qv_g,
    const float* __restrict__ biasr, const float* __restrict__ biasu,
    const float* __restrict__ biasc, const _Float16* __restrict__ wtru,
    const _Float16* __restrict__ wtc, float* __restrict__ out) {
  extern __shared__ char smem[];
  float* sh_h  = (float*)smem;              // 4096
  float* sh_hr = sh_h + NN * DD;            // 4096
  float* sh_u  = sh_hr + NN * DD;           // 4096
  float* sh_m  = sh_u + NN * DD;            // 64
  float* sh_rs = sh_m + NN;                 // 64
  float* sh_qv = sh_rs + NN;                // 320
  _Float16* c_nf = (_Float16*)(sh_qv + 320);        // [64][168]
  _Float16* c_t  = c_nf + NN * CNF_S;               // [160][72]
  _Float16* abf  = c_t + FPAD * CT_S;               // [64][72]
  _Float16* msgb = abf + NN * AB_S;                 // [64][168]

  const int tid = threadIdx.x;
  const int lane = tid & 31;
  const int wave = tid >> 5;
  const int b = blockIdx.x;
  const int len = lengths[b];

  // ---- init: h=0, cache qv, zero the K-padding once --------------------
  for (int i = tid; i < NN * DD; i += 256) sh_h[i] = 0.0f;
  for (int i = tid; i < NN * QQ; i += 256) sh_qv[i] = qv_g[i];
  for (int i = tid; i < NN * (CNF_S - FREAL); i += 256) {
    const int n = i / (CNF_S - FREAL), c = FREAL + i % (CNF_S - FREAL);
    c_nf[n * CNF_S + c] = (_Float16)0.0f;
  }
  for (int i = tid; i < (FPAD - FREAL) * CT_S; i += 256)
    c_t[FREAL * CT_S + i] = (_Float16)0.0f;
  __syncthreads();

  for (int t = 0; t < TT; ++t) {
    const float* xbase = obs + ((size_t)(b * TT + t)) * NN * DD;
    if (t + 1 < TT) __builtin_prefetch(xbase + NN * DD + tid * 16, 0, 0);

    // ---- E1a: masks, rarity scores, combined = [x | rs | h] (both layouts)
    if (tid < NN) {
      const int gi = (b * TT + t) * NN + tid;
      const float mv = mask[gi];
      sh_m[tid] = mv;
      const float r = 0.5f * tanhf(avgint[gi] / (vt[b * NN + tid] + 1.0f));
      sh_rs[tid] = r;
      c_nf[tid * CNF_S + 64] = (_Float16)r;
      c_t[64 * CT_S + tid] = (_Float16)r;
    }
#pragma unroll 4
    for (int k = 0; k < 16; ++k) {
      const int idx = tid + k * 256;
      const int n = idx >> 6, d = idx & 63;
      const float xv = xbase[idx];
      c_nf[n * CNF_S + d] = (_Float16)xv;
      c_t[d * CT_S + n] = (_Float16)xv;
      const float hv = sh_h[idx];
      c_nf[n * CNF_S + 65 + d] = (_Float16)hv;
      c_t[(65 + d) * CT_S + n] = (_Float16)hv;
    }
    __syncthreads();

    // ---- E1b: adjacency A (f16)
#pragma unroll 4
    for (int k = 0; k < 16; ++k) {
      const int idx = tid + k * 256;
      const int i = idx >> 6, j = idx & 63;
      float a;
      if (i == j) a = 1.0f;
      else {
        const float rr = -rarity[idx] * fabsf(sh_rs[i] - sh_rs[j]);
        a = adjnoI[idx] * (1.0f + rr) * sh_m[i] * sh_m[j];
      }
      abf[i * AB_S + j] = (_Float16)a;
    }
    __syncthreads();

    // ---- GEMM1: msg(64x160) = A(64x64) @ combined(64x160), 40 tiles
    for (int it = 0; it < 5; ++it) {
      const int tau = wave + 8 * it;
      const int tm = tau & 3, tn = tau >> 2;
      v8f acc = {};
#pragma unroll
      for (int kc = 0; kc < 2; ++kc) {
        const v16h a = afrag(abf, AB_S, tm * 16, kc * 32, lane);
        const v16h bb = bfrag(c_t + (tn * 16 + (lane & 15)) * CT_S + kc * 32 +
                              ((lane >> 4) << 4));
        acc = wmma32(a, bb, acc);
      }
      const int col = tn * 16 + (lane & 15);
      const int rowb = tm * 16 + ((lane >> 4) << 3);
#pragma unroll
      for (int i = 0; i < 8; ++i)
        msgb[(rowb + i) * MSG_S + col] = (_Float16)acc[i];
    }
    __syncthreads();

    // ---- GEMM2: [r|u](64x128) = sum_q qv[n,q] * (msg @ Wru[q]), 32 tiles
    for (int it = 0; it < 4; ++it) {
      const int tau = wave + 8 * it;
      const int tm = tau & 3, to = tau >> 2;   // to: 0..7 (r:0-3, u:4-7)
      v16h af[5];
#pragma unroll
      for (int kc = 0; kc < 5; ++kc)
        af[kc] = afrag(msgb, MSG_S, tm * 16, kc * 32, lane);
      v8f acc = {};
      const int rowb = tm * 16 + ((lane >> 4) << 3);
#pragma unroll
      for (int q = 0; q < QQ; ++q) {
        const _Float16* wb = wtru +
            ((size_t)(q * 128 + to * 16 + (lane & 15))) * FPAD +
            ((lane >> 4) << 4);
        v8f y = {};
#pragma unroll
        for (int kc = 0; kc < 5; ++kc) y = wmma32(af[kc], bfrag(wb + kc * 32), y);
#pragma unroll
        for (int i = 0; i < 8; ++i) acc[i] += sh_qv[(rowb + i) * QQ + q] * y[i];
      }
      const int col = to * 16 + (lane & 15);
      if (to < 4) {           // r-gate: produce h_r, refresh combined h-slice
#pragma unroll
        for (int i = 0; i < 8; ++i) {
          const int row = rowb + i, o = col, idx = row * 64 + o;
          const float sg = sigm(acc[i] + biasr[idx]);
          const float hv = sh_h[idx];
          const float hrv = (sh_m[row] > 0.0f) ? sg * hv : hv;
          sh_hr[idx] = hrv;
          c_nf[row * CNF_S + 65 + o] = (_Float16)hrv;
          c_t[(65 + o) * CT_S + row] = (_Float16)hrv;
        }
      } else {                // u-gate
#pragma unroll
        for (int i = 0; i < 8; ++i) {
          const int row = rowb + i, o = col - 64;
          sh_u[row * 64 + o] = sigm(acc[i] + biasu[row * 64 + o]);
        }
      }
    }
    __syncthreads();

    // ---- GEMM3: cand(64x64) from comb2, then GRU update, 16 tiles
    for (int it = 0; it < 2; ++it) {
      const int tau = wave + 8 * it;
      const int tm = tau & 3, to = tau >> 2;   // to: 0..3
      v16h af[5];
#pragma unroll
      for (int kc = 0; kc < 5; ++kc)
        af[kc] = afrag(c_nf, CNF_S, tm * 16, kc * 32, lane);
      v8f acc = {};
      const int rowb = tm * 16 + ((lane >> 4) << 3);
#pragma unroll
      for (int q = 0; q < QQ; ++q) {
        const _Float16* wb = wtc +
            ((size_t)(q * 64 + to * 16 + (lane & 15))) * FPAD +
            ((lane >> 4) << 4);
        v8f y = {};
#pragma unroll
        for (int kc = 0; kc < 5; ++kc) y = wmma32(af[kc], bfrag(wb + kc * 32), y);
#pragma unroll
        for (int i = 0; i < 8; ++i) acc[i] += sh_qv[(rowb + i) * QQ + q] * y[i];
      }
      const int col = to * 16 + (lane & 15);
#pragma unroll
      for (int i = 0; i < 8; ++i) {
        const int row = rowb + i, idx = row * 64 + col;
        const float cd = tanhf(acc[i] + biasc[idx]);
        float hnew;
        if (sh_m[row] > 0.0f) {
          const float un = sh_u[idx];
          hnew = (1.0f - un) * sh_hr[idx] + un * cd;
        } else {
          hnew = sh_h[idx];
        }
        sh_h[idx] = hnew;
        if (t == len - 1) out[((size_t)b * NN + row) * DD + col] = hnew;
      }
    }
    __syncthreads();
  }
}

// ---------------------------------------------------------------------------
extern "C" void kernel_launch(void* const* d_in, const int* in_sizes, int n_in,
                              void* d_out, int out_size, void* d_ws,
                              size_t ws_size, hipStream_t stream) {
  (void)in_sizes; (void)n_in; (void)out_size; (void)ws_size;
  const float* obs  = (const float*)d_in[0];
  const float* mask = (const float*)d_in[1];
  const int*   lens = (const int*)d_in[2];
  const float* ai   = (const float*)d_in[3];
  const float* plm  = (const float*)d_in[4];
  const float* rw   = (const float*)d_in[5];
  const float* Wf1  = (const float*)d_in[6];
  const float* bf1  = (const float*)d_in[7];
  const float* Wf2  = (const float*)d_in[8];
  const float* bf2  = (const float*)d_in[9];
  const float* Wg1  = (const float*)d_in[10];
  const float* bg1  = (const float*)d_in[11];
  const float* Wg2  = (const float*)d_in[12];
  const float* bg2  = (const float*)d_in[13];
  const float* Wu   = (const float*)d_in[14];
  const float* bu   = (const float*)d_in[15];
  const float* Wr   = (const float*)d_in[16];
  const float* br   = (const float*)d_in[17];
  const float* Wc   = (const float*)d_in[18];
  const float* bc   = (const float*)d_in[19];

  float* ws = (float*)d_ws;
  float* qv    = ws + WS_QV;
  float* adj   = ws + WS_ADJ;
  float* vt    = ws + WS_VT;
  float* biasr = ws + WS_BR;
  float* biasu = ws + WS_BU;
  float* biasc = ws + WS_BC;
  float* hid   = ws + WS_HID;
  _Float16* wtru = (_Float16*)((char*)d_ws + WS_HALF_BYTE);
  _Float16* wtc  = wtru + WTRU_N;

  vsd_p1a<<<64, 256, 0, stream>>>(plm, Wf1, bf1, Wg1, bg1, hid);
  vsd_p2<<<16, 256, 0, stream>>>(mask, vt);
  vsd_p3<<<600, 256, 0, stream>>>(Wr, Wu, Wc, wtru, wtc);
  vsd_p1b<<<1, 256, 0, stream>>>(hid, Wf2, bf2, Wg2, bg2, qv, adj);
  vsd_p4<<<48, 256, 0, stream>>>(qv, br, bu, bc, biasr, biasu, biasc);

  // 126208 B dynamic LDS (h, hr, u, combined x2, A, msg) -- raise the cap
  const size_t lds_bytes =
      (size_t)(3 * NN * DD + 2 * NN + NN * QQ) * sizeof(float) +
      (size_t)(NN * CNF_S + FPAD * CT_S + NN * AB_S + NN * MSG_S) * sizeof(_Float16);
  (void)hipFuncSetAttribute((const void*)vsd_main,
                            hipFuncAttributeMaxDynamicSharedMemorySize,
                            (int)lds_bytes);
  vsd_main<<<64, 256, lds_bytes, stream>>>(obs, mask, lens, ai, rw, adj, vt, qv,
                                           biasr, biasu, biasc, wtru, wtc,
                                           (float*)d_out);
}
